// BlockMEScaleShift_39307540693472
// MI455X (gfx1250) — compile-verified, matrix operand order checked by reading, not run on hardware
//
#include <hip/hip_runtime.h>
#include <hip/hip_fp16.h>
#include <stdint.h>

typedef __attribute__((ext_vector_type(16))) _Float16 v16h;
typedef __attribute__((ext_vector_type(8)))  _Float16 v8h;
typedef __attribute__((ext_vector_type(4)))  _Float16 v4h;
typedef __attribute__((ext_vector_type(8)))  float    v8f;

#define NPTS   1000000
#define KOFF   27
#define DIM    64
#define DOUT   64
#define TILE_PTS 32            // points per block (two 16-row WMMA tiles)
#define THREADS  128           // 4 wave32
#define LDS_STRIDE 72          // halves per row; 144B stride -> conflict-free b128
#define BUFH (TILE_PTS * LDS_STRIDE)   // halves per gather buffer

// 32-bit LDS byte offset of a generic pointer into a __shared__ object
__device__ __forceinline__ unsigned lds_addr_of(const void* p) {
    return (unsigned)(unsigned long long)
           (__attribute__((address_space(3))) const void*)p;
}

// async global->LDS copy of 32 bytes (two b128 beats), ASYNCcnt-tracked
__device__ __forceinline__ void async_copy_b256(unsigned lds, const void* g) {
    asm volatile("global_load_async_to_lds_b128 %0, %1, off"
                 :: "v"(lds), "v"(g) : "memory");
    asm volatile("global_load_async_to_lds_b128 %0, %1, off offset:16"
                 :: "v"(lds), "v"(g) : "memory");
}

__device__ __forceinline__ void wait_async0() {
    asm volatile("s_wait_asynccnt 0" ::: "memory");
}

// ---------------- prep: W[k][d][o] f32 -> Wt[k][o][d] f16 (transposed) -------
__global__ void wt_convert_kernel(const float* __restrict__ W,
                                  _Float16* __restrict__ Wt) {
    int i = blockIdx.x * blockDim.x + threadIdx.x;      // over K*DOUT*DIM
    if (i >= KOFF * DIM * DOUT) return;
    int k = i / (DIM * DOUT);
    int r = i - k * (DIM * DOUT);
    int o = r / DIM;                                    // out channel
    int d = r - o * DIM;                                // in channel
    Wt[i] = (_Float16)W[k * DIM * DOUT + d * DOUT + o];
}

// ---------------- prep: x f32 -> f16 table (L2-resident gather source) -------
__global__ void x_convert_kernel(const float* __restrict__ x,
                                 _Float16* __restrict__ xh, int n4) {
    int i = blockIdx.x * blockDim.x + threadIdx.x;      // over (N*DIM)/4
    if (i >= n4) return;
    const float4 v = ((const float4*)x)[i];
    v4h h;
    h[0] = (_Float16)v.x; h[1] = (_Float16)v.y;
    h[2] = (_Float16)v.z; h[3] = (_Float16)v.w;
    ((v4h*)xh)[i] = h;
}

// ---------------- main: async gather -> LDS (double buffer) -> WMMA ----------
template<bool WT16, bool X16>
__global__ __launch_bounds__(THREADS)
void spconv_wmma_kernel(const float*   __restrict__ x,
                        const _Float16* __restrict__ xh,
                        const float*   __restrict__ W,
                        const _Float16* __restrict__ Wt,
                        const float*   __restrict__ bias,
                        const float*   __restrict__ scale,
                        const float*   __restrict__ shift,
                        const int*     __restrict__ nidx,
                        const unsigned char* __restrict__ nmask,
                        const int*     __restrict__ bidx,
                        float*         __restrict__ out) {
    __shared__ _Float16 g[2 * BUFH];                    // 9216 B, double buffer

    const int p0      = blockIdx.x * TILE_PTS;
    const int tid     = threadIdx.x;
    const int lane    = tid & 31;
    const int wave    = tid >> 5;
    const int colBase = wave << 4;                      // 16 out-chs per wave
    const int laneLo  = lane & 15;
    const int hi8     = (lane >> 4) << 3;               // K sub-offset 0 / 8

    // gather assignment: 4 threads per point row, 16 halves (32 B) each
    const int gr = tid >> 2;                            // row 0..31
    const int gq = (tid & 3) << 4;                      // half-offset 0,16,32,48

    v8f acc0 = {};                                      // rows p0+0..15
    v8f acc1 = {};                                      // rows p0+16..31

    // issue the gather of offset k into buffer b
    auto issue = [&](int k, int b) {
        const int p  = p0 + gr;
        const int ni = nidx[p * KOFF + k];
        _Float16* dst = &g[b * BUFH + gr * LDS_STRIDE + gq];
        const bool m = nmask[p * KOFF + k] != 0;
        if (X16) {
            if (m) {
                async_copy_b256(lds_addr_of(dst), xh + (size_t)ni * DIM + gq);
            } else {
                v8h z = {};
                *(v8h*)dst = z; *(v8h*)(dst + 8) = z;   // 2x ds_store_b128
            }
        } else {
            v8h h0 = {}, h1 = {};
            if (m) {
                const float* src = x + (size_t)ni * DIM + gq;
                #pragma unroll
                for (int e = 0; e < 8; ++e) h0[e] = (_Float16)src[e];
                #pragma unroll
                for (int e = 0; e < 8; ++e) h1[e] = (_Float16)src[8 + e];
            }
            *(v8h*)dst = h0; *(v8h*)(dst + 8) = h1;
        }
    };

    issue(0, 0);

    for (int k = 0; k < KOFF; ++k) {
        wait_async0();                                  // my gathers landed
        __syncthreads();                                // everyone's landed,
                                                        // prev reads retired
        if (k + 1 < KOFF) issue(k + 1, (k + 1) & 1);

        const _Float16* buf = &g[(k & 1) * BUFH];
        #pragma unroll
        for (int ch = 0; ch < 2; ++ch) {                // two K=32 chunks
            const int kb = (ch << 5) + hi8;
            // B fragment (32x16 f16), shared by both M tiles
            v16h b;
            if (WT16) {
                const _Float16* bp = Wt + (size_t)k * DIM * DOUT
                                        + (size_t)(colBase + laneLo) * DIM + kb;
                v8h blo = *(const v8h*)(bp);            // global_load_b128
                v8h bhi = *(const v8h*)(bp + 16);
                b = __builtin_shufflevector(blo, bhi,
                    0,1,2,3,4,5,6,7,8,9,10,11,12,13,14,15);
            } else {
                const float* wp = W + (size_t)k * DIM * DOUT + colBase + laneLo;
                #pragma unroll
                for (int e = 0; e < 16; ++e) {
                    const int kk = kb + ((e >> 3) << 4) + (e & 7);
                    b[e] = (_Float16)wp[(size_t)kk * DOUT];
                }
            }
            // A fragments: rows 0..15 and 16..31
            const _Float16* a0p = buf + laneLo * LDS_STRIDE + kb;
            const _Float16* a1p = a0p + 16 * LDS_STRIDE;
            v8h a0lo = *(const v8h*)(a0p);              // ds_load_b128 x4
            v8h a0hi = *(const v8h*)(a0p + 16);
            v8h a1lo = *(const v8h*)(a1p);
            v8h a1hi = *(const v8h*)(a1p + 16);
            v16h a0 = __builtin_shufflevector(a0lo, a0hi,
                0,1,2,3,4,5,6,7,8,9,10,11,12,13,14,15);
            v16h a1 = __builtin_shufflevector(a1lo, a1hi,
                0,1,2,3,4,5,6,7,8,9,10,11,12,13,14,15);
            acc0 = __builtin_amdgcn_wmma_f32_16x16x32_f16(
                       false, a0, false, b, (short)0, acc0, false, false);
            acc1 = __builtin_amdgcn_wmma_f32_16x16x32_f16(
                       false, a1, false, b, (short)0, acc1, false, false);
        }
    }

    // epilogue: C layout -> VGPR i, lane<16: M=i ; lane>=16: M=i+8 ; N=lane%16
    const int chO   = colBase + laneLo;
    const float bv  = bias[chO];
    const int rowHi = (lane >> 4) << 3;
    #pragma unroll
    for (int i = 0; i < 8; ++i) {
        const int p  = p0 + i + rowHi;
        const int bi = bidx[p];
        float y = acc0[i] + bv;
        y = y * (scale[bi * DOUT + chO] + 1.0f) + shift[bi * DOUT + chO];
        y = y / (1.0f + __expf(-y));                    // silu
        out[(size_t)p * DOUT + chO] = y;
    }
    #pragma unroll
    for (int i = 0; i < 8; ++i) {
        const int p  = p0 + 16 + i + rowHi;
        const int bi = bidx[p];
        float y = acc1[i] + bv;
        y = y * (scale[bi * DOUT + chO] + 1.0f) + shift[bi * DOUT + chO];
        y = y / (1.0f + __expf(-y));
        out[(size_t)p * DOUT + chO] = y;
    }
}

extern "C" void kernel_launch(void* const* d_in, const int* in_sizes, int n_in,
                              void* d_out, int out_size, void* d_ws, size_t ws_size,
                              hipStream_t stream) {
    const float* x      = (const float*)d_in[0];
    const float* W      = (const float*)d_in[1];
    const float* bias   = (const float*)d_in[2];
    const float* scale  = (const float*)d_in[3];
    const float* shift  = (const float*)d_in[4];
    const int*   nidx   = (const int*)d_in[5];
    const unsigned char* nmask = (const unsigned char*)d_in[6];
    const int*   bidx   = (const int*)d_in[7];
    float* out = (float*)d_out;

    const size_t WT_BYTES = (size_t)KOFF * DIM * DOUT * sizeof(_Float16); // 221184
    const size_t WT_PAD   = (WT_BYTES + 255) & ~(size_t)255;
    const size_t XH_BYTES = (size_t)NPTS * DIM * sizeof(_Float16);        // 128 MB

    const bool useWt = ws_size >= WT_PAD;
    const bool useXh = ws_size >= WT_PAD + XH_BYTES;

    _Float16* Wt = (_Float16*)d_ws;
    _Float16* xh = (_Float16*)((char*)d_ws + WT_PAD);

    if (useWt) {
        const int tot = KOFF * DIM * DOUT;
        wt_convert_kernel<<<(tot + 255) / 256, 256, 0, stream>>>(W, Wt);
    }
    if (useXh) {
        const int n4 = (NPTS * DIM) / 4;
        x_convert_kernel<<<(n4 + 255) / 256, 256, 0, stream>>>(x, xh, n4);
    }

    dim3 grid(NPTS / TILE_PTS);   // 31250, N divides exactly
    if (useXh) {
        spconv_wmma_kernel<true, true><<<grid, THREADS, 0, stream>>>(
            x, xh, W, Wt, bias, scale, shift, nidx, nmask, bidx, out);
    } else if (useWt) {
        spconv_wmma_kernel<true, false><<<grid, THREADS, 0, stream>>>(
            x, xh, W, Wt, bias, scale, shift, nidx, nmask, bidx, out);
    } else {
        spconv_wmma_kernel<false, false><<<grid, THREADS, 0, stream>>>(
            x, xh, W, Wt, bias, scale, shift, nidx, nmask, bidx, out);
    }
}
